// GraphormerModel_21492016349939
// MI455X (gfx1250) — compile-verified, matrix operand order checked by reading, not run on hardware
//
#include <hip/hip_runtime.h>
#include <hip/hip_bf16.h>
#include <math.h>

typedef unsigned short ushort_t;
typedef __attribute__((ext_vector_type(16))) __bf16    v16bf;
typedef __attribute__((ext_vector_type(8)))  float     v8f;
typedef __attribute__((ext_vector_type(8)))  ushort_t  ushort8;

#define NNODES 2048
#define NF_IN  64
#define DMODEL 256
#define NHEAD  8
#define HDIM   32
#define NLAYER 6
#define FFND   1024
#define NEDGE  32768
#define EFD    16
#define NFULL  2049           // N + CLS
#define NP     2080           // padded rows/keys: 65*32, 130*16
#define NEG_BIG (-1.0e30f)

union FragU  { ushort8 u[2]; v16bf bf; };
union Bf16U  { __bf16 b; ushort_t u; };

__device__ __forceinline__ ushort_t f2bf(float x) {
    Bf16U t; t.b = (__bf16)x; return t.u;
}

// Load a 16x32 bf16 WMMA A/B fragment from row-major memory.
// ISA layout (16-bit A 16x32): lanes 0-15 hold rows, K = {0..7,16..23};
// lanes 16-31 hold the same rows, K = {8..15,24..31}.
__device__ __forceinline__ v16bf load_frag_rm(const ushort_t* base, int row0,
                                              int ld, int k0) {
    int lane = threadIdx.x & 31;
    int r = lane & 15, half = lane >> 4;
    const ushort_t* p = base + (size_t)(row0 + r) * ld + k0 + half * 8;
    FragU f;
    f.u[0] = *(const ushort8*)(p);
    f.u[1] = *(const ushort8*)(p + 16);
    return f.bf;
}

// ---------------------------------------------------------------- utilities
__global__ void zero_words(unsigned int* p, int n) {
    int i = blockIdx.x * 256 + threadIdx.x;
    if (i < n) p[i] = 0u;
}

__global__ void f32_to_bf16_k(const float* src, ushort_t* dst, int n) {
    int i = blockIdx.x * 256 + threadIdx.x;
    if (i < n) dst[i] = f2bf(src[i]);
}

// ---------------------------------------------------------------- degrees
__global__ void degrees_k(const int* eidx, int* indeg, int* outdeg) {
    int e = blockIdx.x * 256 + threadIdx.x;
    if (e >= NEDGE) return;
    atomicAdd(&outdeg[eidx[e]], 1);          // edge_index[0] = src
    atomicAdd(&indeg[eidx[NEDGE + e]], 1);   // edge_index[1] = dst
}

// ------------------------------------------------------- node embedding + pad
__global__ void node_embed_k(const float* x, const float* w, const float* b,
                             const float* inEmb, const float* outEmb,
                             const int* indeg, const int* outdeg,
                             const float* cls, float* hbuf) {
    int row = blockIdx.x;           // 0..NP-1
    int d = threadIdx.x;            // 0..255
    if (row == 0) { hbuf[d] = cls[d]; return; }
    if (row > NNODES) { hbuf[(size_t)row * DMODEL + d] = 0.f; return; }
    int i = row - 1;
    float acc = b[d];
    #pragma unroll 8
    for (int k = 0; k < NF_IN; ++k) acc += x[i * NF_IN + k] * w[d * NF_IN + k];
    int di = min(indeg[i], 64), dq = min(outdeg[i], 64);
    acc += inEmb[di * DMODEL + d] + outEmb[dq * DMODEL + d];
    hbuf[(size_t)row * DMODEL + d] = acc;
}

// ------------------------------------------------------------- bias (H,NP,NP)
__global__ void bias_fill_k(const int* dist, const float* dist_bias, float* bias) {
    int idx = blockIdx.x * 256 + threadIdx.x;
    if (idx >= NP * NP) return;
    int i = idx / NP, j = idx % NP;
    if (i < NFULL && j < NFULL) {
        int d0 = (i == 0 || j == 0) ? 0
               : min(max(dist[(i - 1) * NNODES + (j - 1)], 0), 9);
        #pragma unroll
        for (int h = 0; h < NHEAD; ++h)
            bias[((size_t)h * NP + i) * NP + j] = dist_bias[d0 * NHEAD + h];
    } else {
        #pragma unroll
        for (int h = 0; h < NHEAD; ++h)
            bias[((size_t)h * NP + i) * NP + j] = NEG_BIG;  // mask padded keys
    }
}

__global__ void edge_bias_k(const float* eattr, const float* ew, const float* eb,
                            const int* eidx, float* bias) {
    int e = blockIdx.x * 256 + threadIdx.x;
    if (e >= NEDGE) return;
    int src = eidx[e], dst = eidx[NEDGE + e];
    #pragma unroll
    for (int h = 0; h < NHEAD; ++h) {
        float p = eb[h];
        #pragma unroll
        for (int k = 0; k < EFD; ++k) p += eattr[e * EFD + k] * ew[h * EFD + k];
        atomicAdd(&bias[((size_t)h * NP + src + 1) * NP + dst + 1], p);
    }
}

// ------------------------------------------------------------------ layernorm
__global__ void ln_bf16_k(const float* h, const float* s, const float* b,
                          ushort_t* out) {
    int row = blockIdx.x;           // 0..NFULL-1
    int tid = threadIdx.x;          // 0..255
    float v = h[(size_t)row * DMODEL + tid];
    __shared__ float r1[256], r2[256];
    r1[tid] = v; r2[tid] = v * v; __syncthreads();
    for (int st = 128; st > 0; st >>= 1) {
        if (tid < st) { r1[tid] += r1[tid + st]; r2[tid] += r2[tid + st]; }
        __syncthreads();
    }
    float mean = r1[0] * (1.0f / DMODEL);
    float var  = r2[0] * (1.0f / DMODEL) - mean * mean;
    float inv  = rsqrtf(var + 1e-5f);
    out[(size_t)row * DMODEL + tid] = f2bf((v - mean) * inv * s[tid] + b[tid]);
}

__global__ void final_ln_k(const float* h, const float* s, const float* b,
                           float* out) {
    int row = blockIdx.x;
    int tid = threadIdx.x;
    float v = h[(size_t)row * DMODEL + tid];
    __shared__ float r1[256], r2[256];
    r1[tid] = v; r2[tid] = v * v; __syncthreads();
    for (int st = 128; st > 0; st >>= 1) {
        if (tid < st) { r1[tid] += r1[tid + st]; r2[tid] += r2[tid + st]; }
        __syncthreads();
    }
    float mean = r1[0] * (1.0f / DMODEL);
    float var  = r2[0] * (1.0f / DMODEL) - mean * mean;
    float o = (v - mean) * rsqrtf(var + 1e-5f) * s[tid] + b[tid];
    out[(size_t)row * DMODEL + tid] = o;
    if (row == 0) out[(size_t)NFULL * DMODEL + tid] = o;   // second output = h[0]
}

// ------------------------------------------------------------ generic WMMA GEMM
// C = A(M x K, bf16 rm) @ W(N x K, bf16 rm)^T + biasvec
// mode 0: store bf16 row-major (ld = Nout), optional exact GELU
// mode 1: store bf16 transposed out[n*ldout + m]      (for Vt)
// mode 2: res[m*Nout + n] += c  (fp32 residual accumulate)
__global__ __launch_bounds__(32)
void gemm_wmma_k(const ushort_t* A, const ushort_t* W, const float* biasvec,
                 int Nout, int K, int mode, int gelu,
                 ushort_t* outB, int ldout, float* res) {
    int n0 = blockIdx.x * 64;
    int m0 = blockIdx.y * 16;
    int lane = threadIdx.x & 31;
    v8f acc[4] = {};
    for (int k0 = 0; k0 < K; k0 += 32) {
        v16bf a = load_frag_rm(A, m0, K, k0);
        #pragma unroll
        for (int t = 0; t < 4; ++t) {
            v16bf b = load_frag_rm(W, n0 + t * 16, K, k0);
            acc[t] = __builtin_amdgcn_wmma_f32_16x16x32_bf16(
                false, a, false, b, (short)0, acc[t], false, false);
        }
    }
    int col = lane & 15, half = lane >> 4;
    #pragma unroll
    for (int t = 0; t < 4; ++t) {
        int n = n0 + t * 16 + col;
        float bv = biasvec[n];
        #pragma unroll
        for (int r = 0; r < 8; ++r) {
            int m = m0 + r + 8 * half;
            float c = acc[t][r] + bv;
            if (gelu) c = 0.5f * c * (1.0f + erff(c * 0.70710678118f));
            if (mode == 0)      outB[(size_t)m * Nout + n] = f2bf(c);
            else if (mode == 1) outB[(size_t)n * ldout + m] = f2bf(c);
            else                res[(size_t)m * Nout + n] += c;
        }
    }
}

// --------------------------------------------------------- flash attention
// one wave per (head, 16-query tile); 32 keys per step.
__global__ __launch_bounds__(32)
void attn_flash_k(const ushort_t* Qb, const ushort_t* Kb, const ushort_t* Vt,
                  const float* bias, ushort_t* Ob, float scale) {
    int h  = blockIdx.x;
    int q0 = blockIdx.y * 16;
    int lane = threadIdx.x & 31;
    int col = lane & 15, half = lane >> 4;
    __shared__ float Pld[16 * 32];

    v16bf qa = load_frag_rm(Qb, q0, DMODEL, h * HDIM);
    v8f o0 = {}, o1 = {};
    float mrow[8], lrow[8];
    #pragma unroll
    for (int r = 0; r < 8; ++r) { mrow[r] = NEG_BIG; lrow[r] = 0.f; }

    for (int j0 = 0; j0 < NP; j0 += 32) {
        v16bf kb0 = load_frag_rm(Kb, j0,      DMODEL, h * HDIM);
        v16bf kb1 = load_frag_rm(Kb, j0 + 16, DMODEL, h * HDIM);
        v8f z = {};
        v8f s0 = __builtin_amdgcn_wmma_f32_16x16x32_bf16(false, qa, false, kb0,
                                                         (short)0, z, false, false);
        v8f s1 = __builtin_amdgcn_wmma_f32_16x16x32_bf16(false, qa, false, kb1,
                                                         (short)0, z, false, false);
        const float* bp = bias + ((size_t)h * NP + q0) * NP + j0;
        if (j0 + 32 < NP) __builtin_prefetch(bp + 32, 0, 1);   // global_prefetch_b8
        float p0[8], p1[8];
        #pragma unroll
        for (int r = 0; r < 8; ++r) {
            int m = r + 8 * half;
            p0[r] = s0[r] * scale + bp[(size_t)m * NP + col];
            p1[r] = s1[r] * scale + bp[(size_t)m * NP + col + 16];
        }
        #pragma unroll
        for (int r = 0; r < 8; ++r) {
            float mx = fmaxf(p0[r], p1[r]);
            #pragma unroll
            for (int off = 1; off < 16; off <<= 1)
                mx = fmaxf(mx, __shfl_xor(mx, off, 32));
            float mnew  = fmaxf(mrow[r], mx);
            float alpha = __expf(mrow[r] - mnew);
            p0[r] = __expf(p0[r] - mnew);
            p1[r] = __expf(p1[r] - mnew);
            float sum = p0[r] + p1[r];
            #pragma unroll
            for (int off = 1; off < 16; off <<= 1)
                sum += __shfl_xor(sum, off, 32);
            lrow[r] = lrow[r] * alpha + sum;
            mrow[r] = mnew;
            o0[r] *= alpha; o1[r] *= alpha;
            Pld[(r + 8 * half) * 32 + col]      = p0[r];
            Pld[(r + 8 * half) * 32 + col + 16] = p1[r];
        }
        __syncthreads();
        // reload P as a 16x32 bf16 A fragment (lane = query row)
        v16bf pa;
        {
            int m = lane & 15, hh = lane >> 4;
            const float* pr = &Pld[m * 32];
            #pragma unroll
            for (int e = 0; e < 8; ++e) {
                pa[e]     = (__bf16)pr[hh * 8 + e];
                pa[e + 8] = (__bf16)pr[16 + hh * 8 + e];
            }
        }
        v16bf vb0 = load_frag_rm(Vt, h * HDIM,      NP, j0);   // dims 0..15
        v16bf vb1 = load_frag_rm(Vt, h * HDIM + 16, NP, j0);   // dims 16..31
        o0 = __builtin_amdgcn_wmma_f32_16x16x32_bf16(false, pa, false, vb0,
                                                     (short)0, o0, false, false);
        o1 = __builtin_amdgcn_wmma_f32_16x16x32_bf16(false, pa, false, vb1,
                                                     (short)0, o1, false, false);
        __syncthreads();
    }
    #pragma unroll
    for (int r = 0; r < 8; ++r) {
        float inv = 1.0f / lrow[r];
        size_t m = (size_t)(q0 + r + 8 * half);
        Ob[m * DMODEL + h * HDIM + col]      = f2bf(o0[r] * inv);
        Ob[m * DMODEL + h * HDIM + col + 16] = f2bf(o1[r] * inv);
    }
}

// ================================================================== launcher
static inline size_t align256(size_t x) { return (x + 255) & ~(size_t)255; }

extern "C" void kernel_launch(void* const* d_in, const int* in_sizes, int n_in,
                              void* d_out, int out_size, void* d_ws, size_t ws_size,
                              hipStream_t stream) {
    const float* x          = (const float*)d_in[0];
    const float* edge_attr  = (const float*)d_in[1];
    const float* np_w       = (const float*)d_in[2];
    const float* np_b       = (const float*)d_in[3];
    const float* in_emb     = (const float*)d_in[4];
    const float* out_emb    = (const float*)d_in[5];
    const float* dist_bias  = (const float*)d_in[6];
    const float* ep_w       = (const float*)d_in[7];
    const float* ep_b       = (const float*)d_in[8];
    const float* cls        = (const float*)d_in[9];
    const float* q_w        = (const float*)d_in[10];
    const float* q_b        = (const float*)d_in[11];
    const float* k_w        = (const float*)d_in[12];
    const float* k_b        = (const float*)d_in[13];
    const float* v_w        = (const float*)d_in[14];
    const float* v_b        = (const float*)d_in[15];
    const float* o_w        = (const float*)d_in[16];
    const float* o_b        = (const float*)d_in[17];
    const float* f1_w       = (const float*)d_in[18];
    const float* f1_b       = (const float*)d_in[19];
    const float* f2_w       = (const float*)d_in[20];
    const float* f2_b       = (const float*)d_in[21];
    const float* ln1_s      = (const float*)d_in[22];
    const float* ln1_b      = (const float*)d_in[23];
    const float* ln2_s      = (const float*)d_in[24];
    const float* ln2_b      = (const float*)d_in[25];
    const float* fn_s       = (const float*)d_in[26];
    const float* fn_b       = (const float*)d_in[27];
    const int*   edge_index = (const int*)d_in[28];
    const int*   dist_mat   = (const int*)d_in[29];
    float* out = (float*)d_out;

    // ---------------- workspace carve-up
    char* wsb = (char*)d_ws;
    size_t off = 0;
    auto alloc = [&](size_t bytes) -> void* {
        void* p = wsb + off; off = align256(off + bytes); return p;
    };
    float*    bias = (float*)   alloc((size_t)NHEAD * NP * NP * 4);
    float*    hbuf = (float*)   alloc((size_t)NP * DMODEL * 4);
    ushort_t* hn   = (ushort_t*)alloc((size_t)NP * DMODEL * 2);
    ushort_t* Qb   = (ushort_t*)alloc((size_t)NP * DMODEL * 2);
    ushort_t* Kb   = (ushort_t*)alloc((size_t)NP * DMODEL * 2);
    ushort_t* Ob   = (ushort_t*)alloc((size_t)NP * DMODEL * 2);
    ushort_t* Vt   = (ushort_t*)alloc((size_t)DMODEL * NP * 2);
    ushort_t* act  = (ushort_t*)alloc((size_t)NP * FFND * 2);
    ushort_t* wq   = (ushort_t*)alloc((size_t)NLAYER * DMODEL * DMODEL * 2);
    ushort_t* wk   = (ushort_t*)alloc((size_t)NLAYER * DMODEL * DMODEL * 2);
    ushort_t* wv   = (ushort_t*)alloc((size_t)NLAYER * DMODEL * DMODEL * 2);
    ushort_t* wo   = (ushort_t*)alloc((size_t)NLAYER * DMODEL * DMODEL * 2);
    ushort_t* wf1  = (ushort_t*)alloc((size_t)NLAYER * FFND * DMODEL * 2);
    ushort_t* wf2  = (ushort_t*)alloc((size_t)NLAYER * DMODEL * FFND * 2);
    int*      indeg  = (int*)   alloc((size_t)NNODES * 4);
    int*      outdeg = (int*)   alloc((size_t)NNODES * 4);
    (void)ws_size; (void)n_in; (void)in_sizes; (void)out_size;

    // ---------------- setup
    zero_words<<<(2 * NNODES + 255) / 256, 256, 0, stream>>>((unsigned int*)indeg, 2 * NNODES);
    zero_words<<<((int)((size_t)NP * DMODEL * 2 / 4) + 255) / 256, 256, 0, stream>>>(
        (unsigned int*)hn, (int)((size_t)NP * DMODEL * 2 / 4));
    degrees_k<<<(NEDGE + 255) / 256, 256, 0, stream>>>(edge_index, indeg, outdeg);
    node_embed_k<<<NP, 256, 0, stream>>>(x, np_w, np_b, in_emb, out_emb,
                                         indeg, outdeg, cls, hbuf);
    bias_fill_k<<<(NP * NP + 255) / 256, 256, 0, stream>>>(dist_mat, dist_bias, bias);
    edge_bias_k<<<(NEDGE + 255) / 256, 256, 0, stream>>>(edge_attr, ep_w, ep_b,
                                                         edge_index, bias);
    const int nDD = NLAYER * DMODEL * DMODEL, nFD = NLAYER * FFND * DMODEL;
    f32_to_bf16_k<<<(nDD + 255) / 256, 256, 0, stream>>>(q_w, wq, nDD);
    f32_to_bf16_k<<<(nDD + 255) / 256, 256, 0, stream>>>(k_w, wk, nDD);
    f32_to_bf16_k<<<(nDD + 255) / 256, 256, 0, stream>>>(v_w, wv, nDD);
    f32_to_bf16_k<<<(nDD + 255) / 256, 256, 0, stream>>>(o_w, wo, nDD);
    f32_to_bf16_k<<<(nFD + 255) / 256, 256, 0, stream>>>(f1_w, wf1, nFD);
    f32_to_bf16_k<<<(nFD + 255) / 256, 256, 0, stream>>>(f2_w, wf2, nFD);

    const float scale = 0.17677669529663687f;  // 1/sqrt(32)
    dim3 gP(DMODEL / 64, NP / 16);             // projection GEMMs: 4 x 130
    dim3 gF1(FFND / 64, NP / 16);              // FFN1: 16 x 130
    dim3 gA(NHEAD, NP / 16);                   // attention: 8 x 130

    for (int l = 0; l < NLAYER; ++l) {
        const size_t oDD = (size_t)l * DMODEL * DMODEL;
        const size_t oFD = (size_t)l * FFND * DMODEL;
        ln_bf16_k<<<NFULL, 256, 0, stream>>>(hbuf, ln1_s + l * DMODEL,
                                             ln1_b + l * DMODEL, hn);
        gemm_wmma_k<<<gP, 32, 0, stream>>>(hn, wq + oDD, q_b + l * DMODEL,
                                           DMODEL, DMODEL, 0, 0, Qb, 0, nullptr);
        gemm_wmma_k<<<gP, 32, 0, stream>>>(hn, wk + oDD, k_b + l * DMODEL,
                                           DMODEL, DMODEL, 0, 0, Kb, 0, nullptr);
        gemm_wmma_k<<<gP, 32, 0, stream>>>(hn, wv + oDD, v_b + l * DMODEL,
                                           DMODEL, DMODEL, 1, 0, Vt, NP, nullptr);
        attn_flash_k<<<gA, 32, 0, stream>>>(Qb, Kb, Vt, bias, Ob, scale);
        gemm_wmma_k<<<gP, 32, 0, stream>>>(Ob, wo + oDD, o_b + l * DMODEL,
                                           DMODEL, DMODEL, 2, 0, nullptr, 0, hbuf);
        ln_bf16_k<<<NFULL, 256, 0, stream>>>(hbuf, ln2_s + l * DMODEL,
                                             ln2_b + l * DMODEL, hn);
        gemm_wmma_k<<<gF1, 32, 0, stream>>>(hn, wf1 + oFD, f1_b + l * FFND,
                                            FFND, DMODEL, 0, 1, act, 0, nullptr);
        gemm_wmma_k<<<gP, 32, 0, stream>>>(act, wf2 + oFD, f2_b + l * DMODEL,
                                           DMODEL, FFND, 2, 0, nullptr, 0, hbuf);
    }
    final_ln_k<<<NFULL, 256, 0, stream>>>(hbuf, fn_s, fn_b, out);
}
